// Disen_GAT_For_Multi_Aspect_18030272708857
// MI455X (gfx1250) — matrix operand order repaired and checked by reading, not compiled, not run on hardware
//
#include <hip/hip_runtime.h>

typedef __attribute__((ext_vector_type(16))) __bf16 v16bf;
typedef __attribute__((ext_vector_type(8)))  float  v8f;

union FragU {
  v16bf v;
  __bf16 e[16];
  unsigned short s[16];
  uint4 q[2];
};

#define INV_SCALE 0.08838834764831845f  // 1/sqrt(128)

// manual RNE f32->bf16 (used only in tiny one-shot swizzle kernels)
__device__ __forceinline__ unsigned short f2bf(float f){
  union { float f; unsigned u; } x; x.f = f;
  unsigned r = x.u + 0x7FFFu + ((x.u >> 16) & 1u);
  return (unsigned short)(r >> 16);
}

__device__ __forceinline__ v8f vzero8(){
  v8f r;
  #pragma unroll
  for (int i = 0; i < 8; ++i) r[i] = 0.0f;
  return r;
}

__device__ __forceinline__ v8f wmma_bf16(v16bf a, v16bf b, v8f c){
  return __builtin_amdgcn_wmma_f32_16x16x32_bf16(false, a, false, b, (short)0, c, false, false);
}

// A-fragment (16x32 bf16) from global f32, row-major row pointer p, k-chunk base k0
// already including per-lane kbase. Layout: h<8 -> K=k0+h ; h>=8 -> K=k0+16+(h-8).
__device__ __forceinline__ v16bf load_afrag_g(const float* __restrict__ p, int k0){
  FragU u;
  float4 a0 = *(const float4*)(p + k0);
  float4 a1 = *(const float4*)(p + k0 + 4);
  float4 a2 = *(const float4*)(p + k0 + 16);
  float4 a3 = *(const float4*)(p + k0 + 20);
  u.e[0]=(__bf16)a0.x;  u.e[1]=(__bf16)a0.y;  u.e[2]=(__bf16)a0.z;  u.e[3]=(__bf16)a0.w;
  u.e[4]=(__bf16)a1.x;  u.e[5]=(__bf16)a1.y;  u.e[6]=(__bf16)a1.z;  u.e[7]=(__bf16)a1.w;
  u.e[8]=(__bf16)a2.x;  u.e[9]=(__bf16)a2.y;  u.e[10]=(__bf16)a2.z; u.e[11]=(__bf16)a2.w;
  u.e[12]=(__bf16)a3.x; u.e[13]=(__bf16)a3.y; u.e[14]=(__bf16)a3.z; u.e[15]=(__bf16)a3.w;
  return u.v;
}

// A-fragment from an LDS bf16 row (128 halves per row), k0 includes per-lane kbase.
__device__ __forceinline__ v16bf load_afrag_lds(const __bf16* p, int k0){
  FragU u;
  u.q[0] = *(const uint4*)(p + k0);
  u.q[1] = *(const uint4*)(p + k0 + 16);
  return u.v;
}

// ---------------- K0: AS_Q = aspect @ Wq + bq ----------------
__global__ __launch_bounds__(128) void asq_kernel(
    const float* __restrict__ af, const float* __restrict__ Wq,
    const float* __restrict__ bq, float* __restrict__ asq)
{
  int ba = blockIdx.x, h = threadIdx.x;
  float acc = bq[h];
  const float* x = af + (long)ba * 1024;
  for (int d = 0; d < 1024; ++d) acc += x[d] * Wq[d * 128 + h];
  asq[ba * 128 + h] = acc;
}

// ---------------- K_w: swizzle Wk, Wv, W1_W, trans_W into bf16 B-frag layout --------
// B-frag layout for 32x16 bf16 tile: element index = ((kt*8+nt)*32 + lane)*16 + h
// value = W[(kt*32 + (lane/16)*8 + (h<8 ? h : h+8)) * 128 + (nt*16 + lane%16)]
__global__ __launch_bounds__(256) void swz_weights(
    const float* __restrict__ Wk, const float* __restrict__ Wv,
    const float* __restrict__ W1, const float* __restrict__ Tr,
    unsigned short* __restrict__ wkB, unsigned short* __restrict__ wvB,
    unsigned short* __restrict__ w1B, unsigned short* __restrict__ trB)
{
  int e = blockIdx.x * 256 + threadIdx.x;   // 327680 total
  const float* W; unsigned short* o; int idx;
  if (e < 131072)      { W = Wk; o = wkB; idx = e; }
  else if (e < 262144) { W = Wv; o = wvB; idx = e - 131072; }
  else if (e < 294912) { W = W1; o = w1B; idx = e - 262144; }
  else                 { W = Tr; o = trB; idx = e - 294912; }
  int h  = idx & 15;
  int l  = (idx >> 4) & 31;
  int nt = (idx >> 9) & 7;
  int kt = idx >> 12;
  int K   = kt * 32 + ((l >> 4) << 3) + (h < 8 ? h : h + 8);
  int col = nt * 16 + (l & 15);
  o[idx] = f2bf(W[K * 128 + col]);
}

// ---------------- K1: big projections via WMMA bf16 ----------------
// z=0: feature -> W_Km (Wk,bk,mask), W_Vm (Wv,bv,mask)
// z=1: all_type -> T_K (Wk,bk, no mask), T_Vm (Wv,bv,mask)
// z=2: dep      -> DW_Km (Wk,bk,mask)
__global__ __launch_bounds__(256) void proj_kernel(
    const float* __restrict__ feat, const float* __restrict__ allt,
    const float* __restrict__ dep,  const float* __restrict__ fmask,
    const unsigned short* __restrict__ wkB, const unsigned short* __restrict__ wvB,
    const float* __restrict__ bk, const float* __restrict__ bv,
    float* __restrict__ wKm, float* __restrict__ wVm,
    float* __restrict__ tK,  float* __restrict__ tVm, float* __restrict__ dwKm)
{
  const int z = blockIdx.y;
  const float* __restrict__ X = (z == 0) ? feat : ((z == 1) ? allt : dep);
  const int tid  = threadIdx.x;
  const int wave = tid >> 5, lane = tid & 31, l16 = lane & 15;
  const int kb   = (lane < 16) ? 0 : 8;
  const long r0  = (long)blockIdx.x * 128 + wave * 16;
  const int ba   = (int)(r0 >> 9);
  const int b    = ba >> 2;
  const long rowA = r0 + l16;
  const float* __restrict__ xr = X + rowA * 1024;

  v8f acc0[8], acc1[8];
  #pragma unroll
  for (int i = 0; i < 8; ++i) { acc0[i] = vzero8(); acc1[i] = vzero8(); }
  const bool two = (z < 2);

  for (int kt = 0; kt < 32; ++kt) {
    if (kt < 31) __builtin_prefetch(xr + (kt + 1) * 32 + kb, 0, 3);
    v16bf af = load_afrag_g(xr, kt * 32 + kb);
    const unsigned short* wkp = wkB + ((long)kt * 256 + lane) * 16;
    #pragma unroll
    for (int nt = 0; nt < 8; ++nt)
      acc0[nt] = wmma_bf16(af, *(const v16bf*)(wkp + nt * 512), acc0[nt]);
    if (two) {
      const unsigned short* wvp = wvB + ((long)kt * 256 + lane) * 16;
      #pragma unroll
      for (int nt = 0; nt < 8; ++nt)
        acc1[nt] = wmma_bf16(af, *(const v16bf*)(wvp + nt * 512), acc1[nt]);
    }
  }

  const int rhi = (lane >> 4) * 8;
  #pragma unroll
  for (int g = 0; g < 8; ++g) {
    const long rowc = r0 + rhi + g;
    const float m = fmask[b * 512 + (int)(rowc & 511)];
    #pragma unroll
    for (int nt = 0; nt < 8; ++nt) {
      const int col = nt * 16 + l16;
      const float v0 = acc0[nt][g] + bk[col];
      if (z == 0) {
        wKm[rowc * 128 + col] = v0 * m;
        wVm[rowc * 128 + col] = (acc1[nt][g] + bv[col]) * m;
      } else if (z == 1) {
        tK[rowc * 128 + col]  = v0;                       // T_K unmasked
        tVm[rowc * 128 + col] = (acc1[nt][g] + bv[col]) * m;
      } else {
        dwKm[rowc * 128 + col] = v0 * m;
      }
    }
  }
}

// ---------------- K2a: QT[ba][j][k] = sum_i q[i] * T1[i][j][k] ----------------
__global__ __launch_bounds__(256) void qt_kernel(
    const float* __restrict__ asq, const float* __restrict__ T1,
    float* __restrict__ qtF)
{
  __shared__ float qS[128];
  const int ba = blockIdx.x, jc = blockIdx.y;
  const int t = threadIdx.x;
  if (t < 128) qS[t] = asq[ba * 128 + t];
  __syncthreads();
  const int k = t & 127, jh = t >> 7;
  for (int jj = 0; jj < 16; ++jj) {
    const int j = jc * 32 + jh * 16 + jj;
    float acc = 0.0f;
    for (int i = 0; i < 128; ++i)
      acc += qS[i] * T1[(long)i * 16384 + j * 128 + k];
    qtF[((long)ba * 128 + j) * 128 + k] = acc;
  }
}

// ---------------- K2b: swizzle QT into bf16 B-frag layout for B = QT^T ------------
__global__ __launch_bounds__(256) void swz_qt(
    const float* __restrict__ qtF, unsigned short* __restrict__ qtB)
{
  int e = blockIdx.x * 256 + threadIdx.x;  // 524288 total
  int ba = e >> 14, r = e & 16383;
  int h  = r & 15;
  int l  = (r >> 4) & 31;
  int nt = (r >> 9) & 7;
  int kt = r >> 12;
  int kk = kt * 32 + ((l >> 4) << 3) + (h < 8 ? h : h + 8);  // contraction index (H of W_Km)
  int j  = nt * 16 + (l & 15);             // output column
  qtB[e] = f2bf(qtF[((long)ba * 128 + j) * 128 + kk]);
}

// ---------------- K3: fused lp+tp -> ntn -> DW_K2 ----------------
__global__ __launch_bounds__(256) void fuse_kernel(
    const float* __restrict__ asq,  const float* __restrict__ wKm,
    const float* __restrict__ dwKm,
    const unsigned short* __restrict__ w1B, const unsigned short* __restrict__ trB,
    const unsigned short* __restrict__ qtB,
    const float* __restrict__ w1b, const float* __restrict__ trb,
    const float* __restrict__ fmask, float* __restrict__ dwK2)
{
  __shared__ __bf16 ntnS[8 * 16 * 128];  // 32KB, per-wave 16x128 bf16 tile
  __shared__ float qS[128];
  const int tid  = threadIdx.x;
  const int wave = tid >> 5, lane = tid & 31, l16 = lane & 15;
  const int kb   = (lane < 16) ? 0 : 8;
  const long r0  = (long)blockIdx.x * 128 + wave * 16;
  const int ba   = (int)(r0 >> 9);
  const int b    = ba >> 2;
  if (tid < 128) qS[tid] = asq[ba * 128 + tid];
  __syncthreads();

  const long rowA = r0 + l16;
  const float mA = fmask[b * 512 + (int)(rowA & 511)];

  v8f acc[8];
  #pragma unroll
  for (int i = 0; i < 8; ++i) acc[i] = vzero8();

  // part 1: ASQR (asq*mask per row) @ W1_W[0:128]
  #pragma unroll
  for (int kt = 0; kt < 4; ++kt) {
    FragU u;
    #pragma unroll
    for (int h = 0; h < 16; ++h) {
      int K = kt * 32 + kb + (h < 8 ? h : h + 8);
      u.e[h] = (__bf16)(qS[K] * mA);
    }
    const unsigned short* wp = w1B + ((long)kt * 256 + lane) * 16;
    #pragma unroll
    for (int nt = 0; nt < 8; ++nt)
      acc[nt] = wmma_bf16(u.v, *(const v16bf*)(wp + nt * 512), acc[nt]);
  }
  // part 2: W_Km @ W1_W[128:256]  +  W_Km @ QT^T (tp)
  const float* __restrict__ wkr = wKm + rowA * 128;
  const unsigned short* __restrict__ qb0 = qtB + (long)ba * 16384;
  #pragma unroll
  for (int kt = 0; kt < 4; ++kt) {
    v16bf af = load_afrag_g(wkr, kt * 32 + kb);
    const unsigned short* wp = w1B + ((long)(kt + 4) * 256 + lane) * 16;
    const unsigned short* qp = qb0 + ((long)kt * 256 + lane) * 16;
    #pragma unroll
    for (int nt = 0; nt < 8; ++nt) {
      acc[nt] = wmma_bf16(af, *(const v16bf*)(wp + nt * 512), acc[nt]);
      acc[nt] = wmma_bf16(af, *(const v16bf*)(qp + nt * 512), acc[nt]);
    }
  }

  // epilogue: ntn = (acc + W1_b) * mask  -> LDS as bf16 (C-layout -> row major)
  const int rhi = (lane >> 4) * 8;
  __bf16* myS = ntnS + wave * 2048;
  #pragma unroll
  for (int g = 0; g < 8; ++g) {
    const int rloc = rhi + g;
    const float m = fmask[b * 512 + (int)((r0 + rloc) & 511)];
    #pragma unroll
    for (int nt = 0; nt < 8; ++nt) {
      const int col = nt * 16 + l16;
      myS[rloc * 128 + col] = (__bf16)((acc[nt][g] + w1b[col]) * m);
    }
  }
  asm volatile("s_wait_dscnt 0" ::: "memory");  // same-wave LDS RAW

  // GEMM2: DW_K2 = [DW_Km | ntn] @ trans_W + trans_b
  v8f acc2[8];
  #pragma unroll
  for (int i = 0; i < 8; ++i) acc2[i] = vzero8();
  const float* __restrict__ dwr = dwKm + rowA * 128;
  #pragma unroll
  for (int kt = 0; kt < 4; ++kt) {
    v16bf af = load_afrag_g(dwr, kt * 32 + kb);
    const unsigned short* tp = trB + ((long)kt * 256 + lane) * 16;
    #pragma unroll
    for (int nt = 0; nt < 8; ++nt)
      acc2[nt] = wmma_bf16(af, *(const v16bf*)(tp + nt * 512), acc2[nt]);
  }
  const __bf16* lrow = ntnS + wave * 2048 + l16 * 128;
  #pragma unroll
  for (int kt = 0; kt < 4; ++kt) {
    v16bf af = load_afrag_lds(lrow, kt * 32 + kb);
    const unsigned short* tp = trB + ((long)(kt + 4) * 256 + lane) * 16;
    #pragma unroll
    for (int nt = 0; nt < 8; ++nt)
      acc2[nt] = wmma_bf16(af, *(const v16bf*)(tp + nt * 512), acc2[nt]);
  }
  #pragma unroll
  for (int g = 0; g < 8; ++g) {
    const long rowc = r0 + rhi + g;
    #pragma unroll
    for (int nt = 0; nt < 8; ++nt) {
      const int col = nt * 16 + l16;
      dwK2[rowc * 128 + col] = acc2[nt][g] + trb[col];
    }
  }
}

// ---------------- K4: logits + masked softmax x3 + combine + weighted sum ---------
__global__ __launch_bounds__(256) void attn_final(
    const float* __restrict__ asq, const float* __restrict__ tK,
    const float* __restrict__ wKm, const float* __restrict__ dwK2,
    const float* __restrict__ wVm, const float* __restrict__ tVm,
    const float* __restrict__ fmask, const float* __restrict__ comb,
    float* __restrict__ out)
{
  const int ba = blockIdx.x, b = ba >> 2;
  const int t = threadIdx.x;
  __shared__ float qS[128];
  __shared__ float lg[3][512];
  __shared__ float att[512];
  __shared__ float redA[8], redB[8];
  __shared__ float oz[2][128];

  if (t < 128) qS[t] = asq[ba * 128 + t];
  __syncthreads();

  const int rsub = t >> 2, seg = t & 3;
  for (int pass = 0; pass < 8; ++pass) {
    const int n = pass * 64 + rsub;
    const float* qp = qS + seg * 32;
    const long base = ((long)ba * 512 + n) * 128 + seg * 32;
    float s0 = 0.f, s1 = 0.f, s2 = 0.f;
    #pragma unroll
    for (int i = 0; i < 32; ++i) {
      const float qv = qp[i];
      s0 += qv * tK[base + i];
      s1 += qv * wKm[base + i];
      s2 += qv * dwK2[base + i];
    }
    #pragma unroll
    for (int o = 1; o < 4; o <<= 1) {
      s0 += __shfl_xor(s0, o, 4);
      s1 += __shfl_xor(s1, o, 4);
      s2 += __shfl_xor(s2, o, 4);
    }
    if (seg == 0) {
      lg[0][n] = s0 * INV_SCALE;
      lg[1][n] = s1 * INV_SCALE;
      lg[2][n] = s2 * INV_SCALE;
    }
  }
  __syncthreads();

  const float f0 = fmask[b * 512 + t];
  const float f1 = fmask[b * 512 + t + 256];
  att[t] = 0.f; att[t + 256] = 0.f;
  const int wid = t >> 5;
  for (int s = 0; s < 3; ++s) {
    float x0 = lg[s][t], x1 = lg[s][t + 256];
    x0 = f0 * x0 + (1.f - f0) * -1e30f;
    x1 = f1 * x1 + (1.f - f1) * -1e30f;
    float mx = fmaxf(x0, x1);
    #pragma unroll
    for (int o = 16; o > 0; o >>= 1) mx = fmaxf(mx, __shfl_xor(mx, o, 32));
    if ((t & 31) == 0) redA[wid] = mx;
    __syncthreads();
    float gmx = redA[0];
    #pragma unroll
    for (int i = 1; i < 8; ++i) gmx = fmaxf(gmx, redA[i]);
    const float e0 = __expf(x0 - gmx), e1 = __expf(x1 - gmx);
    float sm = e0 + e1;
    #pragma unroll
    for (int o = 16; o > 0; o >>= 1) sm += __shfl_xor(sm, o, 32);
    if ((t & 31) == 0) redB[wid] = sm;
    __syncthreads();
    float gsm = 0.f;
    #pragma unroll
    for (int i = 0; i < 8; ++i) gsm += redB[i];
    const float w = comb[s] / gsm;
    att[t]       += e0 * w;
    att[t + 256] += e1 * w;
    __syncthreads();
  }

  // att_z[h] = sum_n att[n] * W_Vm[n][h] * T_Vm[n][h]
  const int h = t & 127, half = t >> 7;
  float acc = 0.f;
  for (int n = half * 256; n < half * 256 + 256; ++n) {
    const long idx = ((long)ba * 512 + n) * 128 + h;
    acc += att[n] * wVm[idx] * tVm[idx];
  }
  oz[half][h] = acc;
  __syncthreads();
  if (t < 128) out[ba * 128 + t] = oz[0][t] + oz[1][t];
}

extern "C" void kernel_launch(void* const* d_in, const int* in_sizes, int n_in,
                              void* d_out, int out_size, void* d_ws, size_t ws_size,
                              hipStream_t stream)
{
  (void)in_sizes; (void)n_in; (void)out_size; (void)ws_size;
  const float* feature = (const float*)d_in[0];
  const float* dep     = (const float*)d_in[1];
  const float* aspect  = (const float*)d_in[2];
  const float* allt    = (const float*)d_in[3];
  const float* fmask   = (const float*)d_in[4];
  const float* Wq      = (const float*)d_in[5];
  const float* bq      = (const float*)d_in[6];
  const float* Wk      = (const float*)d_in[7];
  const float* bk      = (const float*)d_in[8];
  const float* Wv      = (const float*)d_in[9];
  const float* bv      = (const float*)d_in[10];
  const float* trans_W = (const float*)d_in[11];
  const float* trans_b = (const float*)d_in[12];
  const float* T1      = (const float*)d_in[13];
  const float* W1_W    = (const float*)d_in[14];
  const float* W1_b    = (const float*)d_in[15];
  const float* comb    = (const float*)d_in[16];
  float* out = (float*)d_out;

  char* ws = (char*)d_ws;
  size_t off = 0;
  auto alloc = [&](size_t bytes) -> char* {
    char* p = ws + off;
    off = (off + bytes + 255) & ~(size_t)255;
    return p;
  };
  const size_t BIG = (size_t)16384 * 128 * sizeof(float);  // 8MB
  float* wKm  = (float*)alloc(BIG);
  float* wVm  = (float*)alloc(BIG);
  float* tK   = (float*)alloc(BIG);
  float* tVm  = (float*)alloc(BIG);
  float* dwKm = (float*)alloc(BIG);
  float* dwK2 = (float*)alloc(BIG);
  float* qtF  = (float*)alloc((size_t)32 * 128 * 128 * sizeof(float));
  unsigned short* qtB = (unsigned short*)alloc((size_t)32 * 16384 * 2);
  unsigned short* wkB = (unsigned short*)alloc((size_t)131072 * 2);
  unsigned short* wvB = (unsigned short*)alloc((size_t)131072 * 2);
  unsigned short* w1B = (unsigned short*)alloc((size_t)32768 * 2);
  unsigned short* trB = (unsigned short*)alloc((size_t)32768 * 2);
  float* asqW = (float*)alloc((size_t)4096 * sizeof(float));

  asq_kernel<<<32, 128, 0, stream>>>(aspect, Wq, bq, asqW);
  swz_weights<<<1280, 256, 0, stream>>>(Wk, Wv, W1_W, trans_W, wkB, wvB, w1B, trB);
  dim3 g1(128, 3);
  proj_kernel<<<g1, 256, 0, stream>>>(feature, allt, dep, fmask, wkB, wvB, bk, bv,
                                      wKm, wVm, tK, tVm, dwKm);
  dim3 g2(32, 4);
  qt_kernel<<<g2, 256, 0, stream>>>(asqW, T1, qtF);
  swz_qt<<<2048, 256, 0, stream>>>(qtF, qtB);
  fuse_kernel<<<128, 256, 0, stream>>>(asqW, wKm, dwKm, w1B, trB, qtB,
                                       W1_b, trans_b, fmask, dwK2);
  attn_final<<<32, 256, 0, stream>>>(asqW, tK, wKm, dwK2, wVm, tVm, fmask, comb, out);
}
